// KmeansQuantize_19516331393420
// MI455X (gfx1250) — matrix operand order repaired and checked by reading, not compiled, not run on hardware
//
#include <hip/hip_runtime.h>

// ---------------------------------------------------------------------------
// KmeansQuantize for MI455X (gfx1250): fused GEMM-argmax-gather.
//   similarity[m, v] = sum_k input[m,k] * embd[v,k]     (bf16 WMMA, f32 acc)
//   idx[m]   = argmax_v similarity[m, v]
//   out_q[m] = embd[idx[m], :]                          (f32 gather)
// M = B*T = 32768, D = 256, V = 8192.
// ---------------------------------------------------------------------------

typedef __attribute__((ext_vector_type(16))) __bf16 v16bf;
typedef __attribute__((ext_vector_type(2)))  __bf16 v2bf;
typedef __attribute__((ext_vector_type(8)))  float  v8f;

#define M_ROWS     32768   // B*T
#define D_DIM      256
#define V_DIM      8192
#define STAGE_ROWS 64                      // embd rows staged in LDS per step
#define SB_STRIDE  264                     // 256 + 8 halves pad (bank spread)

#if __has_builtin(__builtin_amdgcn_sched_barrier)
#define SCHED_FENCE() __builtin_amdgcn_sched_barrier(0)
#else
#define SCHED_FENCE()
#endif

__device__ __forceinline__ unsigned int pack2bf(float a, float b) {
#if __has_builtin(__builtin_amdgcn_cvt_pk_bf16_f32)
  v2bf t = __builtin_amdgcn_cvt_pk_bf16_f32(a, b);
  return __builtin_bit_cast(unsigned int, t);
#else
  unsigned int ua = __float_as_uint(a);
  unsigned int ub = __float_as_uint(b);
  ua = (ua + 0x7fffu + ((ua >> 16) & 1u)) >> 16;   // round-to-nearest-even
  ub = (ub + 0x7fffu + ((ub >> 16) & 1u)) >> 16;
  return ua | (ub << 16);
#endif
}

union BF16x16 {
  v16bf v;
  unsigned int u[8];
  uint4 q[2];
};

__global__ __launch_bounds__(256)
void KmeansQuantize_vq_kernel(const float* __restrict__ inp,
                              const float* __restrict__ embd,
                              float* __restrict__ out_q,
                              float* __restrict__ out_idx) {
  __shared__ unsigned short sB[STAGE_ROWS * SB_STRIDE];  // ~33 KB bf16 stage

  const int tid  = threadIdx.x;
  const int lane = tid & 31;
  const int wave = tid >> 5;
  const int half = lane >> 4;     // 0 -> C rows r..r+7 ; 1 -> rows r+8..r+15
  const int l16  = lane & 15;

  const int m0 = (blockIdx.x * 8 + wave) * 16;   // this wave's M-tile base

  // ---- A tile (16 rows x 256 K) -> registers, contiguous k = 16*half + s per chunk ----------
  BF16x16 a[8];
  {
    const float* arow = inp + (size_t)(m0 + l16) * D_DIM;
    #pragma unroll
    for (int c = 0; c < 8; ++c) {
      const int k0 = c * 32 + (half ? 16 : 0);  // halves 0..7
      const int k1 = k0 + 8;                     // halves 8..15
      const float4* p0 = (const float4*)(arow + k0);
      const float4* p1 = (const float4*)(arow + k1);
      float4 f0 = p0[0], f1 = p0[1], f2 = p1[0], f3 = p1[1];
      a[c].u[0] = pack2bf(f0.x, f0.y);  a[c].u[1] = pack2bf(f0.z, f0.w);
      a[c].u[2] = pack2bf(f1.x, f1.y);  a[c].u[3] = pack2bf(f1.z, f1.w);
      a[c].u[4] = pack2bf(f2.x, f2.y);  a[c].u[5] = pack2bf(f2.z, f2.w);
      a[c].u[6] = pack2bf(f3.x, f3.y);  a[c].u[7] = pack2bf(f3.z, f3.w);
    }
  }

  float bestv[8];
  int   besti[8];
  #pragma unroll
  for (int r = 0; r < 8; ++r) { bestv[r] = -3.402823466e38f; besti[r] = 0; }

  // cooperative stage-fill assignment: thread -> (row, 64-col segment)
  const int srow = tid >> 2;
  const int sseg = (tid & 3) * 64;

  // lane's B-fragment base inside a stage (tile t adds t*16*SB_STRIDE rows,
  // k-chunk c adds c*32 halves)
  const unsigned short* bbase = &sB[l16 * SB_STRIDE + (half ? 16 : 0)];

  for (int stage = 0; stage < V_DIM / STAGE_ROWS; ++stage) {
    const int vbase = stage * STAGE_ROWS;

    __syncthreads();   // previous compute finished reading sB

    // ---- fill sB: 64 embd rows, f32 -> bf16, converted once per WG --------
    {
      const float* erow = embd + (size_t)(vbase + srow) * D_DIM + sseg;
      unsigned short* drow = &sB[srow * SB_STRIDE + sseg];
      if (stage + 1 < V_DIM / STAGE_ROWS)
        __builtin_prefetch(erow + STAGE_ROWS * D_DIM, 0, 0);
      #pragma unroll
      for (int i = 0; i < 4; ++i) {            // 16 floats per iter
        const float4* p = (const float4*)(erow + i * 16);
        float4 f0 = p[0], f1 = p[1], f2 = p[2], f3 = p[3];
        uint4 q0, q1;
        q0.x = pack2bf(f0.x, f0.y);  q0.y = pack2bf(f0.z, f0.w);
        q0.z = pack2bf(f1.x, f1.y);  q0.w = pack2bf(f1.z, f1.w);
        q1.x = pack2bf(f2.x, f2.y);  q1.y = pack2bf(f2.z, f2.w);
        q1.z = pack2bf(f3.x, f3.y);  q1.w = pack2bf(f3.z, f3.w);
        uint4* d = (uint4*)(drow + i * 16);
        d[0] = q0;  d[1] = q1;
      }
    }

    __syncthreads();   // sB visible to all waves

    // ---- 4 N-tiles processed together, 1-chunk-lookahead pipeline ---------
    // sched_barrier fences stop the machine scheduler from sinking the
    // lookahead ds_loads back down to their consumers (which would serialize
    // every WMMA behind an s_wait_dscnt 0x0).
    v8f acc[4];
    #pragma unroll
    for (int t = 0; t < 4; ++t) acc[t] = (v8f){0.f,0.f,0.f,0.f,0.f,0.f,0.f,0.f};

    BF16x16 bbuf[2][4];          // double-buffered B fragments (4 tiles each)

    #pragma unroll
    for (int t = 0; t < 4; ++t) {                 // prologue: chunk 0
      const uint4* p = (const uint4*)(bbase + t * (16 * SB_STRIDE));
      bbuf[0][t].q[0] = p[0];
      bbuf[0][t].q[1] = p[1];
    }
    SCHED_FENCE();

    #pragma unroll
    for (int c = 0; c < 8; ++c) {
      const int cur = c & 1;
      const int nxt = cur ^ 1;
      if (c < 7) {                                // issue chunk c+1 loads now
        #pragma unroll
        for (int t = 0; t < 4; ++t) {
          const uint4* p =
              (const uint4*)(bbase + t * (16 * SB_STRIDE) + (c + 1) * 32);
          bbuf[nxt][t].q[0] = p[0];
          bbuf[nxt][t].q[1] = p[1];
        }
      }
      SCHED_FENCE();   // loads for c+1 stay above the chunk-c WMMA group
      #pragma unroll
      for (int t = 0; t < 4; ++t) {               // consume chunk c
        acc[t] = __builtin_amdgcn_wmma_f32_16x16x32_bf16(
            false, a[c].v, false, bbuf[cur][t].v, (short)0, acc[t], false, false);
      }
      SCHED_FENCE();   // WMMAs stay above the next iteration's loads
    }

    #pragma unroll
    for (int t = 0; t < 4; ++t) {
      const int col = vbase + t * 16 + l16;
      #pragma unroll
      for (int r = 0; r < 8; ++r) {
        if (acc[t][r] > bestv[r]) { bestv[r] = acc[t][r]; besti[r] = col; }
      }
    }
  }

  // ---- argmax reduction over the 16 columns held per half-wave -------------
  #pragma unroll
  for (int r = 0; r < 8; ++r) {
    float v = bestv[r];
    int   i = besti[r];
    #pragma unroll
    for (int off = 8; off >= 1; off >>= 1) {
      float ov = __shfl_xor(v, off, 16);
      int   oi = __shfl_xor(i, off, 16);
      if (ov > v || (ov == v && oi < i)) { v = ov; i = oi; }  // argmax, low idx
    }
    bestv[r] = v;
    besti[r] = i;
  }

  // ---- outputs: index (as float) + gathered f32 codebook row ---------------
  #pragma unroll
  for (int r = 0; r < 8; ++r) {
    const int row = m0 + half * 8 + r;     // half 0: rows r; half 1: rows r+8
    const int idx = besti[r];              // uniform across the 16-lane half
    if (l16 == 0) out_idx[row] = (float)idx;
    const float* src = embd + (size_t)idx * D_DIM + l16 * 16;
    float*       dst = out_q + (size_t)row * D_DIM + l16 * 16;
    float4 g0 = ((const float4*)src)[0];
    float4 g1 = ((const float4*)src)[1];
    float4 g2 = ((const float4*)src)[2];
    float4 g3 = ((const float4*)src)[3];
    ((float4*)dst)[0] = g0;  ((float4*)dst)[1] = g1;
    ((float4*)dst)[2] = g2;  ((float4*)dst)[3] = g3;
  }
}

extern "C" void kernel_launch(void* const* d_in, const int* in_sizes, int n_in,
                              void* d_out, int out_size, void* d_ws, size_t ws_size,
                              hipStream_t stream) {
  (void)in_sizes; (void)n_in; (void)out_size; (void)d_ws; (void)ws_size;
  const float* inp  = (const float*)d_in[0];   // [B,T,D] f32
  const float* embd = (const float*)d_in[1];   // [V,D]   f32
  float* out_q   = (float*)d_out;                          // [B,T,D]
  float* out_idx = (float*)d_out + (size_t)M_ROWS * D_DIM; // [B,T]

  dim3 grid(M_ROWS / (16 * 8));   // 256 workgroups, 8 waves x 16 rows each
  dim3 block(256);
  KmeansQuantize_vq_kernel<<<grid, block, 0, stream>>>(inp, embd, out_q, out_idx);
}
